// Model_72696616452480
// MI455X (gfx1250) — matrix operand order repaired and checked by reading, not compiled, not run on hardware
//
#include <hip/hip_runtime.h>
#include <hip/hip_bf16.h>

#define N_NODES 20000
#define N_EDGES 400000
#define HID     64
#define IN_CH   128
#define OUT_CH  40
#define MAX_ITER 15
#define M_AND   5
#define LAM_AND 1e-4f
#define CFAC    0.4f   // (a/(1+a)) * GAMMA = 0.5 * 0.8

typedef __attribute__((ext_vector_type(16))) _Float16 v16h;
typedef __attribute__((ext_vector_type(8)))  float    v8f;

static __device__ __forceinline__ void atomic_add_f32(float* p, float v) {
  __hip_atomic_fetch_add(p, v, __ATOMIC_RELAXED, __HIP_MEMORY_SCOPE_AGENT);
}

// ---------------------------------------------------------------------------
// Cayley transform on-device: Q = (I+S)^{-1}(I-S), S = B - B^T (64x64).
// One block, 64 threads, Gauss-Jordan in LDS. Writes Qh[h][k] = 0.4*Q[h][k] f16.
// ---------------------------------------------------------------------------
__global__ void cayley_kernel(const float* __restrict__ B, _Float16* __restrict__ Qh) {
  __shared__ float M[64][130];
  int tid = threadIdx.x;   // 0..63
  for (int j = 0; j < 64; ++j) {
    float s = B[tid * 64 + j] - B[j * 64 + tid];
    float e = (tid == j) ? 1.f : 0.f;
    M[tid][j]      = e + s;
    M[tid][64 + j] = e - s;
  }
  __syncthreads();
  for (int k = 0; k < 64; ++k) {
    if (tid == k) {
      float ip = 1.f / M[k][k];
      for (int j = k; j < 128; ++j) M[k][j] *= ip;
    }
    __syncthreads();
    if (tid != k) {
      float f = M[tid][k];
      for (int j = k; j < 128; ++j) M[tid][j] -= f * M[k][j];
    }
    __syncthreads();
  }
  for (int j = 0; j < 64; ++j)
    Qh[tid * 64 + j] = (_Float16)(CFAC * M[tid][64 + j]);
}

// Convert Uw (64x128) to f16, Vw (40x64) to f16 padded to 48 rows.
__global__ void conv_weights_kernel(const float* __restrict__ Uw, const float* __restrict__ Vw,
                                    _Float16* __restrict__ Uwh, _Float16* __restrict__ Vwh) {
  int i = blockIdx.x * blockDim.x + threadIdx.x;
  if (i < HID * IN_CH) Uwh[i] = (_Float16)Uw[i];
  if (i < 48 * HID) {
    int o = i >> 6;
    Vwh[i] = (o < OUT_CH) ? (_Float16)Vw[i] : (_Float16)0.f;
  }
}

// ---------------------------------------------------------------------------
// Generic wave-tiled WMMA GEMM: C(MxNT*16) = A(MxK,f32) * Bt^T, Bt is (NT*16 x K) f16
// (Bt[n][k] = B[k][n]).  4 waves/block, 16 rows per wave-tile, K step 32.
// ---------------------------------------------------------------------------
template <int KDIM, int NT>
static __device__ __forceinline__ void gemm_wave(
    const float* __restrict__ A, const _Float16* __restrict__ Bt, int M,
    float* __restrict__ C, int ldc, int colLimit,
    const float* __restrict__ biasVec, float* __restrict__ accAdd) {
  int lane  = threadIdx.x & 31;
  int wave  = threadIdx.x >> 5;
  int mtile = blockIdx.x * (blockDim.x >> 5) + wave;
  if (mtile * 16 >= M) return;
  int kh   = lane >> 4;        // K half selector
  int ncol = lane & 15;        // column within 16-wide tile / A row within tile
  int mrow = mtile * 16 + ncol;

  v8f acc[NT];
#pragma unroll
  for (int nt = 0; nt < NT; ++nt)
#pragma unroll
    for (int r = 0; r < 8; ++r) acc[nt][r] = 0.f;

  for (int kb = 0; kb < KDIM; kb += 32) {
    const float4* Ap = (const float4*)(A + (size_t)mrow * KDIM + kb + kh * 16);
    float4 f0 = Ap[0], f1 = Ap[1], f2 = Ap[2], f3 = Ap[3];
    v16h a;
    a[0]  = (_Float16)f0.x; a[1]  = (_Float16)f0.y; a[2]  = (_Float16)f0.z; a[3]  = (_Float16)f0.w;
    a[4]  = (_Float16)f1.x; a[5]  = (_Float16)f1.y; a[6]  = (_Float16)f1.z; a[7]  = (_Float16)f1.w;
    a[8]  = (_Float16)f2.x; a[9]  = (_Float16)f2.y; a[10] = (_Float16)f2.z; a[11] = (_Float16)f2.w;
    a[12] = (_Float16)f3.x; a[13] = (_Float16)f3.y; a[14] = (_Float16)f3.z; a[15] = (_Float16)f3.w;
#pragma unroll
    for (int nt = 0; nt < NT; ++nt) {
      const _Float16* Br = Bt + (size_t)(nt * 16 + ncol) * KDIM + kb + kh * 16;
      v16h b = *(const v16h*)Br;
      acc[nt] = __builtin_amdgcn_wmma_f32_16x16x32_f16(
          false, a, false, b, (short)0, acc[nt], false, false);
    }
  }
#pragma unroll
  for (int nt = 0; nt < NT; ++nt) {
#pragma unroll
    for (int r = 0; r < 8; ++r) {
      int row = mtile * 16 + kh * 8 + r;
      int col = nt * 16 + ncol;
      if (col < colLimit) {
        float v = acc[nt][r];
        if (biasVec) v += biasVec[col];
        size_t o = (size_t)row * ldc + col;
        C[o] = v;
        if (accAdd) accAdd[o] += v;
      }
    }
  }
}

__global__ void gemm_bias_kernel(const float* A, const _Float16* Bt, const float* bias,
                                 float* C, int M) {
  gemm_wave<IN_CH, 4>(A, Bt, M, C, HID, HID, bias, nullptr);
}
__global__ void gemm_qapply_kernel(const float* A, const _Float16* Bt, float* term,
                                   float* acc, int M) {
  gemm_wave<HID, 4>(A, Bt, M, term, HID, HID, nullptr, acc);
}
__global__ void gemm_out_kernel(const float* A, const _Float16* Bt, float* C, int M) {
  gemm_wave<HID, 3>(A, Bt, M, C, OUT_CH, OUT_CH, nullptr, nullptr);
}

// ---------------------------------------------------------------------------
// Elementwise / sparse kernels
// ---------------------------------------------------------------------------
__global__ void zero4_kernel(float4* p, int n4) {
  int i = blockIdx.x * blockDim.x + threadIdx.x;
  if (i < n4) p[i] = make_float4(0.f, 0.f, 0.f, 0.f);
}

// u12 = |u|; term = acc = u12 + bias   (a = 1.0)
__global__ void ewpre_kernel(const float4* __restrict__ u, const float4* __restrict__ bias,
                             float4* __restrict__ u12, float4* __restrict__ term,
                             float4* __restrict__ acc, int n4) {
  int i = blockIdx.x * blockDim.x + threadIdx.x;
  if (i >= n4) return;
  float4 uu = u[i], bb = bias[i];
  float4 a12 = make_float4(fabsf(uu.x), fabsf(uu.y), fabsf(uu.z), fabsf(uu.w));
  float4 t = make_float4(a12.x + bb.x, a12.y + bb.y, a12.z + bb.z, a12.w + bb.w);
  u12[i] = a12; term[i] = t; acc[i] = t;
}

// spmm: one wave per edge, each lane does 2 channels: agg[dst] += w * Z[src]
__global__ void spmm_kernel(const int* __restrict__ ei, const float* __restrict__ w,
                            const float* __restrict__ Z, float* __restrict__ agg) {
  int gid  = blockIdx.x * blockDim.x + threadIdx.x;
  int e    = gid >> 5;
  int lane = gid & 31;
  if (e >= N_EDGES) return;
  int   src = ei[e];
  int   dst = ei[N_EDGES + e];
  float we  = w[e];
  float2 z  = *(const float2*)(Z + (size_t)src * HID + lane * 2);
  float* o  = agg + (size_t)dst * HID + lane * 2;
  atomic_add_f32(o,     we * z.x);
  atomic_add_f32(o + 1, we * z.y);
}

// g = acc - u12; Fb[i] = g; Xb[i] = u
__global__ void bank_kernel(const float4* __restrict__ acc, const float4* __restrict__ u12,
                            const float4* __restrict__ u, float4* __restrict__ Xb,
                            float4* __restrict__ Fb, int n4, int slot) {
  int i = blockIdx.x * blockDim.x + threadIdx.x;
  if (i >= n4) return;
  float4 a = acc[i], m = u12[i];
  Fb[(size_t)slot * n4 + i] = make_float4(a.x - m.x, a.y - m.y, a.z - m.z, a.w - m.w);
  Xb[(size_t)slot * n4 + i] = u[i];
}

__global__ void zeroH_kernel(float* H) {
  if (threadIdx.x < 32) H[threadIdx.x] = 0.f;
}

// Gram matrix H[j][l] = sum_idx R_j R_l for j<=l<mcount (R = Fb - Xb)
__global__ void hred_kernel(const float* __restrict__ Xb, const float* __restrict__ Fb,
                            float* __restrict__ H, int d, int mcount) {
  float p[15];
#pragma unroll
  for (int t = 0; t < 15; ++t) p[t] = 0.f;
  int stride = gridDim.x * blockDim.x;
  for (int idx = blockIdx.x * blockDim.x + threadIdx.x; idx < d; idx += stride) {
    float r[M_AND];
    for (int j = 0; j < mcount; ++j)
      r[j] = Fb[(size_t)j * d + idx] - Xb[(size_t)j * d + idx];
    int t = 0;
    for (int j = 0; j < mcount; ++j)
      for (int l = j; l < mcount; ++l)
        p[t++] += r[j] * r[l];
  }
  int lane = threadIdx.x & 31;
  int t = 0;
  for (int j = 0; j < mcount; ++j)
    for (int l = j; l < mcount; ++l) {
      float v = p[t++];
      for (int m = 16; m >= 1; m >>= 1) v += __shfl_xor(v, m);
      if (lane == 0) atomic_add_f32(&H[j * M_AND + l], v);
    }
}

// Solve (H + lam I) alpha = mask, normalize to sum 1. Single thread (5x5).
__global__ void solve5_kernel(const float* __restrict__ H, float* __restrict__ alpha,
                              int mcount) {
  if (threadIdx.x != 0 || blockIdx.x != 0) return;
  float A[M_AND][M_AND + 1];
  for (int j = 0; j < M_AND; ++j) {
    for (int l = 0; l < M_AND; ++l) {
      float v = (j <= l) ? H[j * M_AND + l] : H[l * M_AND + j];
      A[j][l] = v + ((j == l) ? LAM_AND : 0.f);
    }
    A[j][M_AND] = (j < mcount) ? 1.f : 0.f;
  }
  for (int k = 0; k < M_AND; ++k) {
    int p = k; float best = fabsf(A[k][k]);
    for (int r = k + 1; r < M_AND; ++r) {
      float v = fabsf(A[r][k]);
      if (v > best) { best = v; p = r; }
    }
    if (p != k)
      for (int c = 0; c <= M_AND; ++c) { float t = A[k][c]; A[k][c] = A[p][c]; A[p][c] = t; }
    float ip = 1.f / A[k][k];
    for (int c = k; c <= M_AND; ++c) A[k][c] *= ip;
    for (int r = 0; r < M_AND; ++r)
      if (r != k) {
        float f = A[r][k];
        for (int c = k; c <= M_AND; ++c) A[r][c] -= f * A[k][c];
      }
  }
  float s = 0.f;
  for (int j = 0; j < M_AND; ++j) s += A[j][M_AND];
  float inv = 1.f / s;
  for (int j = 0; j < M_AND; ++j) alpha[j] = A[j][M_AND] * inv;
}

// u = sum_j alpha[j] * Fb[j]
__global__ void mix_kernel(float4* __restrict__ u, const float4* __restrict__ Fb,
                           const float* __restrict__ alpha, int n4) {
  int i = blockIdx.x * blockDim.x + threadIdx.x;
  if (i >= n4) return;
  float a0 = alpha[0], a1 = alpha[1], a2 = alpha[2], a3 = alpha[3], a4 = alpha[4];
  float4 f0 = Fb[i], f1 = Fb[(size_t)n4 + i], f2 = Fb[2 * (size_t)n4 + i],
         f3 = Fb[3 * (size_t)n4 + i], f4 = Fb[4 * (size_t)n4 + i];
  float4 o;
  o.x = a0 * f0.x + a1 * f1.x + a2 * f2.x + a3 * f3.x + a4 * f4.x;
  o.y = a0 * f0.y + a1 * f1.y + a2 * f2.y + a3 * f3.y + a4 * f4.y;
  o.z = a0 * f0.z + a1 * f1.z + a2 * f2.z + a3 * f3.z + a4 * f4.z;
  o.w = a0 * f0.w + a1 * f1.w + a2 * f2.w + a3 * f3.w + a4 * f4.w;
  u[i] = o;
}

// zn[node] = relu(u[node]) / max(||relu(u[node])||, 1e-12); one wave per node.
__global__ void normalize_kernel(const float* __restrict__ u, float* __restrict__ zn) {
  int gid  = blockIdx.x * blockDim.x + threadIdx.x;
  int node = gid >> 5;
  int lane = gid & 31;
  if (node >= N_NODES) return;
  float2 uu = *(const float2*)(u + (size_t)node * HID + lane * 2);
  float z0 = fmaxf(uu.x, 0.f), z1 = fmaxf(uu.y, 0.f);
  float ss = z0 * z0 + z1 * z1;
  for (int m = 16; m >= 1; m >>= 1) ss += __shfl_xor(ss, m);
  float inv = 1.f / fmaxf(sqrtf(ss), 1e-12f);
  *(float2*)(zn + (size_t)node * HID + lane * 2) = make_float2(z0 * inv, z1 * inv);
}

// ---------------------------------------------------------------------------
extern "C" void kernel_launch(void* const* d_in, const int* in_sizes, int n_in,
                              void* d_out, int out_size, void* d_ws, size_t ws_size,
                              hipStream_t stream) {
  const float* x   = (const float*)d_in[0];
  const int*   ei  = (const int*)d_in[1];
  const float* ew  = (const float*)d_in[2];
  const float* Uw  = (const float*)d_in[3];
  const float* Ub  = (const float*)d_in[4];
  const float* Bm  = (const float*)d_in[5];
  const float* Vw  = (const float*)d_in[6];
  float*       out = (float*)d_out;

  const size_t d  = (size_t)N_NODES * HID;   // 1,280,000
  const int    n4 = (int)(d / 4);            // 320,000

  float* bias = (float*)d_ws;
  float* u    = bias + d;
  float* u12  = u + d;
  float* acc  = u12 + d;
  float* term = acc + d;
  float* agg  = term + d;
  float* zn   = agg + d;
  float* Xb   = zn + d;
  float* Fb   = Xb + (size_t)M_AND * d;
  float* Hb   = Fb + (size_t)M_AND * d;          // 32 floats
  float* alp  = Hb + 32;                         // 32 floats
  _Float16* Qh  = (_Float16*)(alp + 32);         // 64*64
  _Float16* Uwh = Qh + 64 * 64;                  // 64*128
  _Float16* Vwh = Uwh + 64 * 128;                // 48*64

  const int EW_BLK   = (n4 + 255) / 256;               // 1250
  const int SPMM_BLK = (N_EDGES * 32 + 255) / 256;     // 50000
  const int MT       = N_NODES / 16;                   // 1250 wave tiles
  const int GEMM_BLK = (MT + 3) / 4;                   // 313
  const int NORM_BLK = (N_NODES * 32 + 255) / 256;     // 2500

  // --- Setup: Cayley Q (f16, scaled by c*gamma), f16 weights, bias GEMM, u = 0
  cayley_kernel<<<1, 64, 0, stream>>>(Bm, Qh);
  conv_weights_kernel<<<32, 256, 0, stream>>>(Uw, Vw, Uwh, Vwh);
  gemm_bias_kernel<<<GEMM_BLK, 128, 0, stream>>>(x, Uwh, Ub, bias, N_NODES);
  zero4_kernel<<<EW_BLK, 256, 0, stream>>>((float4*)u, n4);

  for (int k = 0; k < MAX_ITER; ++k) {
    // u12 = |u|; term = acc = u12 + bias
    ewpre_kernel<<<EW_BLK, 256, 0, stream>>>((const float4*)u, (const float4*)bias,
                                             (float4*)u12, (float4*)term, (float4*)acc, n4);
    // 3 Neumann steps: term = (c*gamma*Q) @ spmm(term); acc += term
    for (int s = 0; s < 3; ++s) {
      zero4_kernel<<<EW_BLK, 256, 0, stream>>>((float4*)agg, n4);
      spmm_kernel<<<SPMM_BLK, 256, 0, stream>>>(ei, ew, term, agg);
      gemm_qapply_kernel<<<GEMM_BLK, 128, 0, stream>>>(agg, Qh, term, acc, N_NODES);
    }
    // Anderson banks + Gram + 5x5 solve + mix
    int slot = k % M_AND;
    int mcnt = (k + 1 < M_AND) ? (k + 1) : M_AND;
    bank_kernel<<<EW_BLK, 256, 0, stream>>>((const float4*)acc, (const float4*)u12,
                                            (const float4*)u, (float4*)Xb, (float4*)Fb,
                                            n4, slot);
    zeroH_kernel<<<1, 32, 0, stream>>>(Hb);
    hred_kernel<<<256, 256, 0, stream>>>(Xb, Fb, Hb, (int)d, mcnt);
    solve5_kernel<<<1, 1, 0, stream>>>(Hb, alp, mcnt);
    mix_kernel<<<EW_BLK, 256, 0, stream>>>((float4*)u, (const float4*)Fb, alp, n4);
  }

  // Output: normalize rows of relu(u), project with Vw (padded 48 cols, 40 kept)
  normalize_kernel<<<NORM_BLK, 256, 0, stream>>>(u, zn);
  gemm_out_kernel<<<GEMM_BLK, 128, 0, stream>>>(zn, Vwh, out, N_NODES);
}
